// AttentionLayer_46488726012573
// MI455X (gfx1250) — matrix operand order repaired and checked by reading, not compiled, not run on hardware
//
#include <hip/hip_runtime.h>
#include <hip/hip_bf16.h>

// ---------------------------------------------------------------------------
// Bahdanau additive attention, fused for MI455X (gfx1250, wave32, WMMA).
//   B=8, Te=512, Td=128, D_ENC=D_DEC=512, ATT=256
// Pipeline:
//   1) pre-split all GEMM operands into bf16 hi + bf16 residual lo arrays
//      (transposes fused with the split where a [N,K] layout is needed)
//   2) Whi = values @ W_h   (WMMA bf16 split GEMM, 16x64 tile/wave)
//   3) Us  = query  @ U_a   (WMMA bf16 split GEMM)
//   4) fused: scores = V . tanh(Whi[s] + Us[t]); softmax -> e (+ bf16 split)
//   5) c = e @ values       (batched WMMA bf16 split GEMM)
// The [B,Td,Te,ATT] activation tensor of the reference (537 MB of traffic)
// never exists; total HBM traffic ~30 MB at 23.3 TB/s.
// ---------------------------------------------------------------------------

typedef __attribute__((ext_vector_type(16))) __bf16 v16bf;
typedef __attribute__((ext_vector_type(8)))  __bf16 v8bf;
typedef __attribute__((ext_vector_type(4)))  __bf16 v4bf;
typedef __attribute__((ext_vector_type(8)))  float  v8f;

#define NB   8
#define NTE  512
#define NTD  128
#define NDE  512
#define NATT 256

__device__ __forceinline__ void split1(float x, __bf16& h, __bf16& l) {
  h = (__bf16)x;
  l = (__bf16)(x - (float)h);
}

// ---------------------------------------------------------------------------
// Elementwise f32 -> bf16 hi/lo split (n multiple of 4).
// ---------------------------------------------------------------------------
__global__ __launch_bounds__(256) void split_kernel(
    const float* __restrict__ in, __bf16* __restrict__ hi,
    __bf16* __restrict__ lo, long n) {
  const long base = ((long)blockIdx.x * blockDim.x + threadIdx.x) * 4;
  if (base >= n) return;
  float4 x = *(const float4*)(in + base);
  float f[4] = {x.x, x.y, x.z, x.w};
  v4bf hv, lv;
#pragma unroll
  for (int k = 0; k < 4; ++k) {
    __bf16 h, l;
    split1(f[k], h, l);
    hv[k] = h; lv[k] = l;
  }
  *(v4bf*)(hi + base) = hv;
  *(v4bf*)(lo + base) = lv;
}

// ---------------------------------------------------------------------------
// Tiled transpose fused with bf16 split: in[b][R][C] -> hi/lo[b][C][R].
// R, C multiples of 32; blockDim = (32,8).
// ---------------------------------------------------------------------------
__global__ __launch_bounds__(256) void transpose_split_kernel(
    const float* __restrict__ in, __bf16* __restrict__ hi,
    __bf16* __restrict__ lo, int R, int C) {
  __shared__ float tile[32][33];
  const int b  = blockIdx.z;
  const float* ib = in + (long)b * R * C;
  const long  obase = (long)b * R * C;
  const int c0 = blockIdx.x * 32, r0 = blockIdx.y * 32;
  const int tx = threadIdx.x, ty = threadIdx.y;
#pragma unroll
  for (int j = 0; j < 32; j += 8)
    tile[ty + j][tx] = ib[(long)(r0 + ty + j) * C + c0 + tx];
  __syncthreads();
#pragma unroll
  for (int j = 0; j < 32; j += 8) {
    const float x = tile[tx][ty + j];
    __bf16 h, l;
    split1(x, h, l);
    const long idx = obase + (long)(c0 + ty + j) * R + r0 + tx;
    hi[idx] = h;
    lo[idx] = l;
  }
}

// ---------------------------------------------------------------------------
// Split-precision bf16 WMMA GEMM on pre-split operands:
//   C[M,N](f32) = (Ahi+Alo)[M,K] * (Bhi+Blo)[N,K]^T, dropping lo*lo.
// One wave computes a 16x64 tile (4 accumulators); A hi/lo fragments are
// loaded once per 32-K step and reused across 4 B sub-tiles -> 12 WMMAs per
// 20 b128 loads. Optional batch via blockIdx.y with flat element strides.
// ---------------------------------------------------------------------------
__global__ __launch_bounds__(256) void gemm_bf16_wmma_kernel(
    const __bf16* __restrict__ Ahi, const __bf16* __restrict__ Alo,
    const __bf16* __restrict__ Bhi, const __bf16* __restrict__ Blo,
    float* __restrict__ C, int M, int N, int K,
    long sA, long sB, long sC) {
  const int lane = threadIdx.x & 31;
  const int wv   = threadIdx.x >> 5;
  const int tiles_n = N >> 6;                  // 64-wide N tiles
  const int tiles   = (M >> 4) * tiles_n;
  const int tile = blockIdx.x * 8 + wv;        // wave-uniform
  if (tile >= tiles) return;                   // whole wave exits together

  const long bo = blockIdx.y;
  const __bf16* ah = Ahi + bo * sA;
  const __bf16* al = Alo + bo * sA;
  const __bf16* bh = Bhi + bo * sB;
  const __bf16* bl = Blo + bo * sB;
  float*        Cb = C   + bo * sC;

  const int tm   = (tile / tiles_n) << 4;
  const int tn   = (tile % tiles_n) << 6;
  const int l15  = lane & 15;
  const int hi16 = lane >> 4;
  const int kbA  = hi16 << 3;                  // A: K-groups of 8 (ISA layout)
  const int kbB  = hi16 << 3;                  // B: K-groups of 8 (ISA layout)

  const __bf16* aph = ah + (long)(tm + l15) * K;
  const __bf16* apl = al + (long)(tm + l15) * K;

  v8f acc[4] = {};
  for (int k0 = 0; k0 < K; k0 += 32) {
    v16bf Ah, Al;
    ((v8bf*)&Ah)[0] = *(const v8bf*)(aph + k0 + kbA);
    ((v8bf*)&Ah)[1] = *(const v8bf*)(aph + k0 + kbA + 16);
    ((v8bf*)&Al)[0] = *(const v8bf*)(apl + k0 + kbA);
    ((v8bf*)&Al)[1] = *(const v8bf*)(apl + k0 + kbA + 16);
#pragma unroll
    for (int j = 0; j < 4; ++j) {
      const long boff = (long)(tn + j * 16 + l15) * K + k0 + kbB;
      v16bf Bh, Bl;
      ((v8bf*)&Bh)[0] = *(const v8bf*)(bh + boff);
      ((v8bf*)&Bh)[1] = *(const v8bf*)(bh + boff + 16);
      ((v8bf*)&Bl)[0] = *(const v8bf*)(bl + boff);
      ((v8bf*)&Bl)[1] = *(const v8bf*)(bl + boff + 16);
      acc[j] = __builtin_amdgcn_wmma_f32_16x16x32_bf16(false, Ah, false, Bh,
                                                       (short)0, acc[j], false, false);
      acc[j] = __builtin_amdgcn_wmma_f32_16x16x32_bf16(false, Ah, false, Bl,
                                                       (short)0, acc[j], false, false);
      acc[j] = __builtin_amdgcn_wmma_f32_16x16x32_bf16(false, Al, false, Bh,
                                                       (short)0, acc[j], false, false);
    }
  }

  // C 16x16 f32 layout: VGPR r -> M = tm + r + (lane>>4)*8, N = col + (lane&15)
  const int mrow = tm + (hi16 << 3);
#pragma unroll
  for (int j = 0; j < 4; ++j)
#pragma unroll
    for (int r = 0; r < 8; ++r)
      Cb[(long)(mrow + r) * N + tn + j * 16 + l15] = acc[j][r];
}

// ---------------------------------------------------------------------------
// Fused scores + softmax + bf16 split of e.  One block per (b,t) pair.
// scores[s] = sum_a V[a] * tanh(Whi[b,s,a] + Us[b,t,a]); softmax over s.
// ---------------------------------------------------------------------------
__global__ __launch_bounds__(256) void scores_softmax_kernel(
    const float* __restrict__ Whi, const float* __restrict__ Us,
    const float* __restrict__ Va, float* __restrict__ e_out,
    __bf16* __restrict__ eHi, __bf16* __restrict__ eLo) {
  const int bt   = blockIdx.x;           // b*NTD + t
  const int b    = bt >> 7;              // NTD = 128
  const int tid  = threadIdx.x;
  const int lane = tid & 31;
  const int wv   = tid >> 5;

  __shared__ float sc[NTE];
  __shared__ float red[8];

  float u[8], vv[8];
  const float* Urow = Us + (long)bt * NATT;
#pragma unroll
  for (int i = 0; i < 8; ++i) {
    const int a = lane + 32 * i;
    u[i]  = Urow[a];
    vv[i] = Va[a];
  }

  for (int s = wv; s < NTE; s += 8) {
    const float* w = Whi + ((long)b * NTE + s) * NATT;
    float p = 0.f;
#pragma unroll
    for (int i = 0; i < 8; ++i)
      p += vv[i] * tanhf(w[lane + 32 * i] + u[i]);
#pragma unroll
    for (int off = 16; off > 0; off >>= 1)
      p += __shfl_down(p, off, 32);
    if (lane == 0) sc[s] = p;
  }
  __syncthreads();

  const float x0 = sc[tid], x1 = sc[tid + 256];
  float vmax = fmaxf(x0, x1);
#pragma unroll
  for (int off = 16; off > 0; off >>= 1)
    vmax = fmaxf(vmax, __shfl_down(vmax, off, 32));
  if (lane == 0) red[wv] = vmax;
  __syncthreads();
  float m = red[0];
#pragma unroll
  for (int i = 1; i < 8; ++i) m = fmaxf(m, red[i]);

  const float e0 = __expf(x0 - m), e1 = __expf(x1 - m);
  float psum = e0 + e1;
#pragma unroll
  for (int off = 16; off > 0; off >>= 1)
    psum += __shfl_down(psum, off, 32);
  __syncthreads();                       // red reuse
  if (lane == 0) red[wv] = psum;
  __syncthreads();
  float tot = 0.f;
#pragma unroll
  for (int i = 0; i < 8; ++i) tot += red[i];
  const float inv = 1.f / tot;

  const float g0 = e0 * inv, g1 = e1 * inv;
  const long row = (long)bt * NTE;
  e_out[row + tid]       = g0;
  e_out[row + tid + 256] = g1;
  __bf16 h, l;
  split1(g0, h, l); eHi[row + tid]       = h; eLo[row + tid]       = l;
  split1(g1, h, l); eHi[row + tid + 256] = h; eLo[row + tid + 256] = l;
}

// ---------------------------------------------------------------------------
extern "C" void kernel_launch(void* const* d_in, const int* in_sizes, int n_in,
                              void* d_out, int out_size, void* d_ws, size_t ws_size,
                              hipStream_t stream) {
  (void)in_sizes; (void)n_in; (void)out_size; (void)ws_size;
  const float* values = (const float*)d_in[0];   // [B, Te, D_ENC]
  const float* query  = (const float*)d_in[1];   // [B, Td, D_DEC]
  const float* W_h    = (const float*)d_in[2];   // [D_ENC, ATT]
  const float* U_a    = (const float*)d_in[3];   // [D_DEC, ATT]
  const float* V_a    = (const float*)d_in[4];   // [1, ATT]

  // ---- workspace layout (16B-aligned chunks) ----
  char* p = (char*)d_ws;
  float* Whi  = (float*)p;  p += sizeof(float) * (NB * NTE) * NATT;   // 4 MB
  float* Us   = (float*)p;  p += sizeof(float) * (NB * NTD) * NATT;   // 1 MB
  __bf16* vAh = (__bf16*)p; p += 2L * (NB * NTE) * NDE;   // values as A (hi)
  __bf16* vAl = (__bf16*)p; p += 2L * (NB * NTE) * NDE;   //              (lo)
  __bf16* qAh = (__bf16*)p; p += 2L * (NB * NTD) * NDE;   // query as A
  __bf16* qAl = (__bf16*)p; p += 2L * (NB * NTD) * NDE;
  __bf16* WBh = (__bf16*)p; p += 2L * NATT * NDE;         // W_h^T as B
  __bf16* WBl = (__bf16*)p; p += 2L * NATT * NDE;
  __bf16* UBh = (__bf16*)p; p += 2L * NATT * NDE;         // U_a^T as B
  __bf16* UBl = (__bf16*)p; p += 2L * NATT * NDE;
  __bf16* VBh = (__bf16*)p; p += 2L * NB * NDE * NTE;     // values^T as B
  __bf16* VBl = (__bf16*)p; p += 2L * NB * NDE * NTE;
  __bf16* eAh = (__bf16*)p; p += 2L * NB * NTD * NTE;     // e as A
  __bf16* eAl = (__bf16*)p; p += 2L * NB * NTD * NTE;

  float* c_out = (float*)d_out;                  // [B, Td, D_ENC]
  float* e_out = c_out + NB * NTD * NDE;         // [B, Td, Te]

  const dim3 tb(32, 8, 1);

  // 1) operand preparation: split (and transpose-split) once, reuse many times
  split_kernel<<<dim3((NB * NTE * NDE) / 4 / 256, 1, 1), 256, 0, stream>>>(
      values, vAh, vAl, (long)NB * NTE * NDE);
  split_kernel<<<dim3((NB * NTD * NDE) / 4 / 256, 1, 1), 256, 0, stream>>>(
      query, qAh, qAl, (long)NB * NTD * NDE);
  transpose_split_kernel<<<dim3(NATT / 32, NDE / 32, 1),  tb, 0, stream>>>(
      W_h, WBh, WBl, NDE, NATT);
  transpose_split_kernel<<<dim3(NATT / 32, NDE / 32, 1),  tb, 0, stream>>>(
      U_a, UBh, UBl, NDE, NATT);
  transpose_split_kernel<<<dim3(NDE / 32, NTE / 32, NB), tb, 0, stream>>>(
      values, VBh, VBl, NTE, NDE);

  // 2) Whi = values @ W_h : M=4096, N=256, K=512 -> (256 m)*(4 n64) = 1024 tiles
  gemm_bf16_wmma_kernel<<<dim3(128, 1, 1), 256, 0, stream>>>(
      vAh, vAl, WBh, WBl, Whi, NB * NTE, NATT, NDE, 0, 0, 0);
  // 3) Us = query @ U_a : M=1024, N=256, K=512 -> 256 tiles
  gemm_bf16_wmma_kernel<<<dim3(32, 1, 1), 256, 0, stream>>>(
      qAh, qAl, UBh, UBl, Us, NB * NTD, NATT, NDE, 0, 0, 0);

  // 4) fused tanh-scores + softmax -> e_out (+ pre-split e for GEMM 5)
  scores_softmax_kernel<<<dim3(NB * NTD, 1, 1), 256, 0, stream>>>(
      Whi, Us, V_a, e_out, eAh, eAl);

  // 5) c = e @ values : batched (8), M=128, N=512, K=512 -> 64 tiles/batch
  gemm_bf16_wmma_kernel<<<dim3(8, NB, 1), 256, 0, stream>>>(
      eAh, eAl, VBh, VBl, c_out, NTD, NDE, NTE,
      (long)NTD * NTE, (long)NDE * NTE, (long)NTD * NDE);
}
